// ReferenceTransformerBlock_71966472012551
// MI455X (gfx1250) — matrix operand order repaired and checked by reading, not compiled, not run on hardware
//
#include <hip/hip_runtime.h>

typedef __bf16 bf16;
typedef __attribute__((ext_vector_type(4)))  __bf16 v4bf;
typedef __attribute__((ext_vector_type(8)))  __bf16 v8bf;
typedef __attribute__((ext_vector_type(16))) __bf16 v16bf;
typedef __attribute__((ext_vector_type(8)))  float  v8f;
typedef __attribute__((ext_vector_type(4)))  int    v4i;

union V16U { v16bf v; v8bf h[2]; };

#define HIDDEN   1024
#define HEADS    16
#define HEAD_DIM 64
#define BATCH    2
#define SEQ      2048
#define ROWS     (BATCH * SEQ)   // 4096

// ---- async global->LDS support (gfx1250), guarded so compile never breaks ----
#if defined(__has_builtin)
#if __has_builtin(__builtin_amdgcn_global_load_async_to_lds_b128) && \
    __has_builtin(__builtin_amdgcn_s_wait_asynccnt)
#define ASYNC_LDS 1
#endif
#endif
#ifndef ASYNC_LDS
#define ASYNC_LDS 0
#endif

#define AS_GLOBAL __attribute__((address_space(1)))
#define AS_LOCAL  __attribute__((address_space(3)))

__device__ __forceinline__ void wait_async_done()
{
#if ASYNC_LDS
    __builtin_amdgcn_s_wait_asynccnt(0);
#endif
}

// ---------------------------------------------------------------------------
// 1) fp32 -> bf16 conversion (vectorized x4)
// ---------------------------------------------------------------------------
__global__ void cvt_f32_to_bf16(const float* __restrict__ in,
                                bf16* __restrict__ out, int n4)
{
    int i = blockIdx.x * blockDim.x + threadIdx.x;
    if (i < n4) {
        float4 v = ((const float4*)in)[i];
        v4bf o;
        o[0] = (bf16)v.x; o[1] = (bf16)v.y; o[2] = (bf16)v.z; o[3] = (bf16)v.w;
        ((v4bf*)out)[i] = o;
    }
}

// ---------------------------------------------------------------------------
// 2) GEMM: Out = A @ W^T  with A [ROWS,HIDDEN], W [HIDDEN,HIDDEN], both bf16,
//    K contiguous in both operands. Output written in [B,H,S,HD] bf16 layout.
//    Block = 256 threads (8 waves) -> 128x128 tile; wave -> 32x64 subtile.
// ---------------------------------------------------------------------------
__global__ void __launch_bounds__(256)
gemm_qkv_bf16(const bf16* __restrict__ A,
              const bf16* __restrict__ W,
              bf16* __restrict__ Out)
{
    const int lane = threadIdx.x & 31;
    const int wave = threadIdx.x >> 5;
    const int lm   = lane & 15;
    const int lhi  = lane >> 4;          // 0 or 1 (half-wave)

    const int m0 = blockIdx.x * 128 + (wave >> 1) * 32;
    const int n0 = blockIdx.y * 128 + (wave & 1) * 64;

    v8f acc[2][4];
    v8f zero = {};
#pragma unroll
    for (int mi = 0; mi < 2; ++mi)
#pragma unroll
        for (int ni = 0; ni < 4; ++ni) acc[mi][ni] = zero;

    for (int k0 = 0; k0 < HIDDEN; k0 += 32) {
        // A fragments (16x32): lane holds row m, K run of 16 starting at lhi*16
        v16bf af[2];
#pragma unroll
        for (int mi = 0; mi < 2; ++mi) {
            const bf16* ap = A + (size_t)(m0 + mi * 16 + lm) * HIDDEN + k0 + lhi * 16;
            V16U u;
            u.h[0] = *(const v8bf*)ap;
            u.h[1] = *(const v8bf*)(ap + 8);
            af[mi] = u.v;
            // speculative prefetch of next K-step (unconditional: OOB prefetch
            // is silently dropped, keeps the hot loop branch-free)
            __builtin_prefetch(ap + 32, 0, 1);
        }
        // B fragments (32x16): lane holds col n, K run of 16 starting at lhi*16
        v16bf bfrag[4];
#pragma unroll
        for (int ni = 0; ni < 4; ++ni) {
            const bf16* bp = W + (size_t)(n0 + ni * 16 + lm) * HIDDEN + k0 + lhi * 16;
            bfrag[ni] = *(const v16bf*)bp;
        }
#pragma unroll
        for (int mi = 0; mi < 2; ++mi)
#pragma unroll
            for (int ni = 0; ni < 4; ++ni)
                acc[mi][ni] = __builtin_amdgcn_wmma_f32_16x16x32_bf16(
                    false, af[mi], false, bfrag[ni], (short)0, acc[mi][ni], false, false);
    }

    // Epilogue: C layout -> [B,H,S,HD] bf16
#pragma unroll
    for (int mi = 0; mi < 2; ++mi)
#pragma unroll
        for (int ni = 0; ni < 4; ++ni)
#pragma unroll
            for (int r = 0; r < 8; ++r) {
                int m = m0 + mi * 16 + r + lhi * 8;
                int n = n0 + ni * 16 + lm;
                int b = m >> 11, s = m & 2047;
                int h = n >> 6,  d = n & 63;
                Out[((size_t)(b * HEADS + h) * SEQ + s) * HEAD_DIM + d] =
                    (bf16)acc[mi][ni][r];
            }
}

// ---------------------------------------------------------------------------
// 3) Flash attention. Block = 128 threads (4 waves) handles 64 queries of one
//    (b,h). Each wave owns 16 query rows. K / V^T tiles double-buffered in
//    LDS; K tile staged with async global->LDS (ASYNCcnt) when available.
// ---------------------------------------------------------------------------
__device__ __forceinline__ void stage_tile(const bf16* __restrict__ Kb,
                                           const bf16* __restrict__ Vb,
                                           size_t base, int kbase,
                                           bf16* ldsKbuf, bf16* ldsVTbuf,
                                           int tid)
{
    for (int i = tid; i < 512; i += 128) {
        int key = i >> 3, dg = (i & 7) * 8;
        const size_t g = base + (size_t)(kbase + key) * HEAD_DIM + dg;
#if ASYNC_LDS
        __builtin_amdgcn_global_load_async_to_lds_b128(
            (AS_GLOBAL v4i*)(AS_GLOBAL void*)(void*)(Kb + g),
            (AS_LOCAL  v4i*)(AS_LOCAL  void*)(void*)(ldsKbuf + key * 64 + dg),
            0, 0);
#else
        v8bf kv = *(const v8bf*)(Kb + g);
        *(v8bf*)(ldsKbuf + key * 64 + dg) = kv;
#endif
        v8bf vv = *(const v8bf*)(Vb + g);
#pragma unroll
        for (int e = 0; e < 8; ++e) ldsVTbuf[(dg + e) * 64 + key] = vv[e];
    }
}

__global__ void __launch_bounds__(128)
flash_attn_bf16(const bf16* __restrict__ Qb, const bf16* __restrict__ Kb,
                const bf16* __restrict__ Vb, float* __restrict__ Out)
{
    const int bh = blockIdx.x;             // b*HEADS + h
    const int qt = blockIdx.y;             // query tile (64 rows)
    const int b  = bh >> 4, h = bh & 15;
    const int tid  = threadIdx.x;
    const int lane = tid & 31;
    const int wave = tid >> 5;             // 0..3
    const int lm   = lane & 15;
    const int lhi  = lane >> 4;
    const size_t base = (size_t)bh * SEQ * HEAD_DIM;
    const float scale = 0.125f;            // 1/sqrt(64)
    const int NT = SEQ / 64;

    __shared__ __align__(16) bf16 ldsK [2][64 * 64];    // [buf][key][dim]
    __shared__ __align__(16) bf16 ldsVT[2][64 * 64];    // [buf][dim][key]
    __shared__ __align__(16) bf16 ldsP [4][16 * 64];    // per-wave P scratch

    // Persistent Q A-fragments (16 rows x 64 dims -> two 16x32 frags)
    v16bf qf[2];
    {
        const bf16* qp = Qb + base + (size_t)(qt * 64 + wave * 16 + lm) * HEAD_DIM;
#pragma unroll
        for (int kk = 0; kk < 2; ++kk) {
            V16U u;
            u.h[0] = *(const v8bf*)(qp + kk * 32 + lhi * 16);
            u.h[1] = *(const v8bf*)(qp + kk * 32 + lhi * 16 + 8);
            qf[kk] = u.v;
        }
    }

    v8f acc[4];
    v8f zero = {};
#pragma unroll
    for (int nt = 0; nt < 4; ++nt) acc[nt] = zero;
    float mrun[8], lrun[8];
#pragma unroll
    for (int r = 0; r < 8; ++r) { mrun[r] = -3.0e38f; lrun[r] = 0.f; }

    // prologue: stage tile 0 into buffer 0
    stage_tile(Kb, Vb, base, 0, &ldsK[0][0], &ldsVT[0][0], tid);

    for (int kt = 0; kt < NT; ++kt) {
        const int cur = kt & 1;
        // current buffer's async copies (issued by this wave) must land,
        // then barrier publishes all waves' LDS writes.
        wait_async_done();
        __syncthreads();

        // kick off next tile into the other buffer (its previous readers
        // finished before the barrier above)
        if (kt + 1 < NT)
            stage_tile(Kb, Vb, base, (kt + 1) * 64,
                       &ldsK[cur ^ 1][0], &ldsVT[cur ^ 1][0], tid);

        // ---- S = Q K^T (16 x 64 per wave) ----
        v8f sf[4];
#pragma unroll
        for (int nt = 0; nt < 4; ++nt) {
            v8f s = zero;
            const int keyn = nt * 16 + lm;
#pragma unroll
            for (int kk = 0; kk < 2; ++kk) {
                v16bf bfm = *(const v16bf*)&ldsK[cur][keyn * 64 + kk * 32 + lhi * 16];
                s = __builtin_amdgcn_wmma_f32_16x16x32_bf16(
                        false, qf[kk], false, bfm, (short)0, s, false, false);
            }
            sf[nt] = s;
        }

        // ---- online softmax (rows striped across 16-lane halves) ----
        float mnew[8];
#pragma unroll
        for (int r = 0; r < 8; ++r) mnew[r] = mrun[r];
#pragma unroll
        for (int nt = 0; nt < 4; ++nt)
#pragma unroll
            for (int r = 0; r < 8; ++r) {
                float v = sf[nt][r] * scale;
                sf[nt][r] = v;
                mnew[r] = fmaxf(mnew[r], v);
            }
#pragma unroll
        for (int r = 0; r < 8; ++r) {
            float v = mnew[r];
#pragma unroll
            for (int xm = 8; xm >= 1; xm >>= 1)
                v = fmaxf(v, __shfl_xor(v, xm, 32));
            mnew[r] = v;
        }
        float rowsum[8];
#pragma unroll
        for (int r = 0; r < 8; ++r) rowsum[r] = 0.f;
#pragma unroll
        for (int nt = 0; nt < 4; ++nt)
#pragma unroll
            for (int r = 0; r < 8; ++r) {
                float p = __expf(sf[nt][r] - mnew[r]);
                sf[nt][r] = p;
                rowsum[r] += p;
            }
#pragma unroll
        for (int r = 0; r < 8; ++r) {
            float v = rowsum[r];
#pragma unroll
            for (int xm = 8; xm >= 1; xm >>= 1)
                v += __shfl_xor(v, xm, 32);
            float alpha = __expf(mrun[r] - mnew[r]);
            lrun[r] = lrun[r] * alpha + v;
            mrun[r] = mnew[r];
#pragma unroll
            for (int nt = 0; nt < 4; ++nt) acc[nt][r] *= alpha;
        }

        // ---- P: C layout -> LDS row-major -> A fragments (wave-local) ----
#pragma unroll
        for (int nt = 0; nt < 4; ++nt)
#pragma unroll
            for (int r = 0; r < 8; ++r)
                ldsP[wave][(r + lhi * 8) * 64 + nt * 16 + lm] = (bf16)sf[nt][r];

        v16bf pf[2];
#pragma unroll
        for (int kk = 0; kk < 2; ++kk) {
            const bf16* pp = &ldsP[wave][lm * 64 + kk * 32 + lhi * 16];
            V16U u;
            u.h[0] = *(const v8bf*)pp;
            u.h[1] = *(const v8bf*)(pp + 8);
            pf[kk] = u.v;
        }

        // ---- O += P V ----
#pragma unroll
        for (int nt = 0; nt < 4; ++nt) {
            const int dn = nt * 16 + lm;
#pragma unroll
            for (int kk = 0; kk < 2; ++kk) {
                v16bf vfm = *(const v16bf*)&ldsVT[cur][dn * 64 + kk * 32 + lhi * 16];
                acc[nt] = __builtin_amdgcn_wmma_f32_16x16x32_bf16(
                              false, pf[kk], false, vfm, (short)0, acc[nt], false, false);
            }
        }
    }

    // ---- epilogue: normalize and write [B,S,D] fp32 ----
#pragma unroll
    for (int r = 0; r < 8; ++r) {
        const float rl = 1.0f / lrun[r];
        const int srow = qt * 64 + wave * 16 + r + lhi * 8;
#pragma unroll
        for (int nt = 0; nt < 4; ++nt) {
            const int dcol = h * HEAD_DIM + nt * 16 + lm;
            Out[(size_t)(b * SEQ + srow) * HIDDEN + dcol] = acc[nt][r] * rl;
        }
    }
}

// ---------------------------------------------------------------------------
// 4) residual + LayerNorm, one block (256 threads) per row of 1024
// ---------------------------------------------------------------------------
__global__ void __launch_bounds__(256)
residual_ln(const float* __restrict__ attn, const float* __restrict__ x,
            const float* __restrict__ gamma, const float* __restrict__ beta,
            float* __restrict__ out)
{
    const int row = blockIdx.x, tid = threadIdx.x;
    const size_t base = (size_t)row * HIDDEN;
    float4 av = ((const float4*)(attn + base))[tid];
    float4 xv = ((const float4*)(x    + base))[tid];
    float4 hv = make_float4(av.x + xv.x, av.y + xv.y, av.z + xv.z, av.w + xv.w);

    float s  = hv.x + hv.y + hv.z + hv.w;
    float ss = hv.x * hv.x + hv.y * hv.y + hv.z * hv.z + hv.w * hv.w;
#pragma unroll
    for (int m = 16; m >= 1; m >>= 1) {
        s  += __shfl_xor(s,  m, 32);
        ss += __shfl_xor(ss, m, 32);
    }
    __shared__ float rs[8], rss[8];
    if ((tid & 31) == 0) { rs[tid >> 5] = s; rss[tid >> 5] = ss; }
    __syncthreads();
    s = 0.f; ss = 0.f;
#pragma unroll
    for (int i = 0; i < 8; ++i) { s += rs[i]; ss += rss[i]; }

    const float mean = s  * (1.0f / HIDDEN);
    const float var  = ss * (1.0f / HIDDEN) - mean * mean;
    const float rstd = rsqrtf(var + 1e-5f);

    float4 g  = ((const float4*)gamma)[tid];
    float4 bb = ((const float4*)beta)[tid];
    float4 o  = make_float4((hv.x - mean) * rstd * g.x + bb.x,
                            (hv.y - mean) * rstd * g.y + bb.y,
                            (hv.z - mean) * rstd * g.z + bb.z,
                            (hv.w - mean) * rstd * g.w + bb.w);
    ((float4*)(out + base))[tid] = o;
}

// ---------------------------------------------------------------------------
// launch
// ---------------------------------------------------------------------------
extern "C" void kernel_launch(void* const* d_in, const int* in_sizes, int n_in,
                              void* d_out, int out_size, void* d_ws, size_t ws_size,
                              hipStream_t stream)
{
    const float* x     = (const float*)d_in[0];
    const float* Wq    = (const float*)d_in[1];
    const float* Wk    = (const float*)d_in[2];
    const float* Wv    = (const float*)d_in[3];
    const float* gamma = (const float*)d_in[4];
    const float* beta  = (const float*)d_in[5];
    float* out = (float*)d_out;

    // workspace carve-out (~54 MB total)
    char* ws = (char*)d_ws;
    size_t off = 0;
    auto carve = [&](size_t bytes) -> char* {
        char* p = ws + off;
        off += (bytes + 255) & ~(size_t)255;
        return p;
    };
    bf16*  xb   = (bf16*) carve((size_t)ROWS * HIDDEN * 2);
    bf16*  wqb  = (bf16*) carve((size_t)HIDDEN * HIDDEN * 2);
    bf16*  wkb  = (bf16*) carve((size_t)HIDDEN * HIDDEN * 2);
    bf16*  wvb  = (bf16*) carve((size_t)HIDDEN * HIDDEN * 2);
    bf16*  Qb   = (bf16*) carve((size_t)ROWS * HIDDEN * 2);
    bf16*  Kb   = (bf16*) carve((size_t)ROWS * HIDDEN * 2);
    bf16*  Vb   = (bf16*) carve((size_t)ROWS * HIDDEN * 2);
    float* attn = (float*)carve((size_t)ROWS * HIDDEN * 4);

    // 1) fp32 -> bf16
    {
        int n4x = ROWS * HIDDEN / 4;
        int n4w = HIDDEN * HIDDEN / 4;
        cvt_f32_to_bf16<<<n4x / 256, 256, 0, stream>>>(x,  xb,  n4x);
        cvt_f32_to_bf16<<<n4w / 256, 256, 0, stream>>>(Wq, wqb, n4w);
        cvt_f32_to_bf16<<<n4w / 256, 256, 0, stream>>>(Wk, wkb, n4w);
        cvt_f32_to_bf16<<<n4w / 256, 256, 0, stream>>>(Wv, wvb, n4w);
    }

    // 2) QKV projections
    dim3 ggrid(ROWS / 128, HIDDEN / 128);
    gemm_qkv_bf16<<<ggrid, 256, 0, stream>>>(xb, wqb, Qb);
    gemm_qkv_bf16<<<ggrid, 256, 0, stream>>>(xb, wkb, Kb);
    gemm_qkv_bf16<<<ggrid, 256, 0, stream>>>(xb, wvb, Vb);

    // 3) flash attention
    dim3 agrid(BATCH * HEADS, SEQ / 64);
    flash_attn_bf16<<<agrid, 128, 0, stream>>>(Qb, Kb, Vb, attn);

    // 4) residual + LayerNorm
    residual_ln<<<ROWS, 256, 0, stream>>>(attn, x, gamma, beta, out);
}
